// SAMN_TRUST_74431783239695
// MI455X (gfx1250) — matrix operand order repaired and checked by reading, not compiled, not run on hardware
//
#include <hip/hip_runtime.h>
#include <hip/hip_bf16.h>

#define EDIM 128
#define MDIM 8
#define ADIM 64
#define NFR  50
#define FM   (NFR * MDIM)   // 400
#define F2STR 130           // padded LDS row stride (floats), even for float2 loads
#define WAFRAG_DW (4 * 4 * 32 * 8)   // nt * kb * lane * dword = 4096 dwords (16KB)

typedef __attribute__((ext_vector_type(16))) __bf16 v16bf;
typedef __attribute__((ext_vector_type(2)))  __bf16 v2bf;
typedef __attribute__((ext_vector_type(8)))  float  v8f;

union Frag { unsigned u[8]; uint4 q[2]; v16bf v; };

// Hardware bf16 convert (RNE) -> packed dword; lowers to v_cvt_pk_bf16_f32.
__device__ __forceinline__ unsigned pack_bf16(float lo, float hi) {
  v2bf p;
  p.x = (__bf16)lo;
  p.y = (__bf16)hi;
  return __builtin_bit_cast(unsigned, p);
}

// Immediate-pattern ds_swizzle: offset[14:10]=xor, [9:5]=or, [4:0]=and.
template <int PAT>
__device__ __forceinline__ float swz(float v) {
  return __int_as_float(__builtin_amdgcn_ds_swizzle(__float_as_int(v), PAT));
}

// Full 32-lane sum (xor butterfly 1,2,4,8,16) — one ds_swizzle per step.
__device__ __forceinline__ float waveSum(float v) {
  v += swz<0x041f>(v);
  v += swz<0x081f>(v);
  v += swz<0x101f>(v);
  v += swz<0x201f>(v);
  v += swz<0x401f>(v);
  return v;
}

// Sum within each 16-lane half (xor 1,2,4,8).
__device__ __forceinline__ float halfSum16(float v) {
  v += swz<0x041f>(v);
  v += swz<0x081f>(v);
  v += swz<0x101f>(v);
  v += swz<0x201f>(v);
  return v;
}

// ---- Kernel 0: pre-pack WA (f32 [128][64]) into WMMA B-fragment order ------
// Layout: dword index = ((nt*4 + kb)*32 + lane)*8 + d
__global__ __launch_bounds__(256) void k_prepWA(
    const float* __restrict__ WA, unsigned* __restrict__ WAbf)
{
  int idx = blockIdx.x * 256 + threadIdx.x;   // 0 .. 4095
  int d    = idx & 7;
  int lane = (idx >> 3) & 31;
  int kb   = (idx >> 8) & 3;
  int nt   = idx >> 10;
  int col  = nt * 16 + (lane & 15);
  int k0   = kb * 32 + ((d < 4) ? 0 : 16) + (lane >> 4) * 8 + 2 * (d & 3);
  WAbf[idx] = pack_bf16(WA[k0 * ADIM + col], WA[(k0 + 1) * ADIM + col]);
}

// ---------------- Kernel 1: att_key[b][f*8+m] = (uid_n * fe_n) @ Key --------
__global__ __launch_bounds__(256) void k_attkey(
    const int* __restrict__ input_u, const int* __restrict__ input_uf,
    const float* __restrict__ uidW, const float* __restrict__ Key,
    float* __restrict__ att, int user_num)
{
  __shared__ float s_keyT[MDIM * EDIM];  // transposed [m][e]
  __shared__ float s_uidn[EDIM];
  const int t = threadIdx.x, lane = t & 31, w = t >> 5;
  const int b = blockIdx.x;

  for (int i = t; i < MDIM * EDIM; i += 256) {
    int m = i >> 7, e = i & 127;
    s_keyT[i] = Key[e * MDIM + m];
  }
  if (w == 0) {
    const float4* up = (const float4*)(uidW + (size_t)input_u[b] * EDIM);
    float4 u4 = up[lane];
    float n2 = waveSum(u4.x*u4.x + u4.y*u4.y + u4.z*u4.z + u4.w*u4.w);
    float inv = 1.0f / fmaxf(sqrtf(n2), 1e-12f);
    ((float4*)s_uidn)[lane] = make_float4(u4.x*inv, u4.y*inv, u4.z*inv, u4.w*inv);
  }
  __syncthreads();

  const float4 un4 = ((const float4*)s_uidn)[lane];
  const int e0 = lane * 4;
  for (int f = w; f < NFR; f += 8) {
    int uf = input_uf[b * NFR + f];
    float msk = (uf != user_num) ? 1.0f : 0.0f;
    float4 fe = ((const float4*)(uidW + (size_t)uf * EDIM))[lane];
    fe.x *= msk; fe.y *= msk; fe.z *= msk; fe.w *= msk;
    float n2 = waveSum(fe.x*fe.x + fe.y*fe.y + fe.z*fe.z + fe.w*fe.w);
    float inv = 1.0f / fmaxf(sqrtf(n2), 1e-12f);
    float c0 = un4.x * fe.x * inv, c1 = un4.y * fe.y * inv;
    float c2 = un4.z * fe.z * inv, c3 = un4.w * fe.w * inv;
    float s[MDIM];
#pragma unroll
    for (int m = 0; m < MDIM; ++m) {
      const float* kr = s_keyT + m * EDIM + e0;
      s[m] = c0*kr[0] + c1*kr[1] + c2*kr[2] + c3*kr[3];
    }
#pragma unroll
    for (int m = 0; m < MDIM; ++m) s[m] = waveSum(s[m]);
    if (lane == 0) {
      float* p = att + (size_t)b * FM + f * MDIM;
#pragma unroll
      for (int m = 0; m < MDIM; ++m) p[m] = s[m];
    }
  }
}

// ---------------- Kernel 2: softmax-over-batch stats per (f,m) --------------
__global__ __launch_bounds__(256) void k_stats(
    const float* __restrict__ att, float* __restrict__ mx,
    float* __restrict__ ise, int Bsz)
{
  __shared__ float red[256];
  const int fm = blockIdx.x, t = threadIdx.x;
  float m = -3.402823466e38f;
  for (int b = t; b < Bsz; b += 256) m = fmaxf(m, att[(size_t)b * FM + fm]);
  red[t] = m; __syncthreads();
  for (int s = 128; s > 0; s >>= 1) {
    if (t < s) red[t] = fmaxf(red[t], red[t + s]);
    __syncthreads();
  }
  float mxv = red[0]; __syncthreads();
  float sum = 0.0f;
  for (int b = t; b < Bsz; b += 256) sum += expf(att[(size_t)b * FM + fm] - mxv);
  red[t] = sum; __syncthreads();
  for (int s = 128; s > 0; s >>= 1) {
    if (t < s) red[t] += red[t + s];
    __syncthreads();
  }
  if (t == 0) { mx[fm] = mxv; ise[fm] = 1.0f / red[0]; }
}

// ---------------- Kernel 3: f2, WMMA GEMM (h), j, friend, score -------------
__global__ __launch_bounds__(256) void k_main(
    const int* __restrict__ input_u, const int* __restrict__ input_i,
    const int* __restrict__ input_uf, const float* __restrict__ uidW,
    const float* __restrict__ iidW, const float* __restrict__ i_bias,
    const float* __restrict__ Mem, const unsigned* __restrict__ WAbf,
    const float* __restrict__ BA, const float* __restrict__ Uo,
    const float* __restrict__ att, const float* __restrict__ mx,
    const float* __restrict__ ise, float* __restrict__ out, int user_num)
{
  __shared__ float s_f2[64 * F2STR];     // f2 rows, padded to 64 rows
  __shared__ float s_mem[MDIM * EDIM];   // Mem [m][e]
  __shared__ float s_uid[EDIM];
  __shared__ float s_attrow[FM];
  __shared__ float s_mx[FM];
  __shared__ float s_ise[FM];
  __shared__ float s_fn[64];
  __shared__ float s_dotp[64 * 2];       // [row][nHalf] deterministic partials
  __shared__ float s_j[64];
  __shared__ float s_tmp[256];
  __shared__ float s_red[128];
  __shared__ float s_sj;

  const int t = threadIdx.x, lane = t & 31, w = t >> 5;
  const int b = blockIdx.x;

  for (int i = t; i < MDIM * EDIM; i += 256) s_mem[i] = Mem[i];
  for (int i = t; i < FM; i += 256) {
    s_attrow[i] = att[(size_t)b * FM + i];
    s_mx[i] = mx[i];
    s_ise[i] = ise[i];
  }
  if (t < EDIM) s_uid[t] = uidW[(size_t)input_u[b] * EDIM + t];
  if (t < 64) s_fn[t] = 0.0f;
  if (t < 128) s_dotp[t] = 0.0f;
  for (int i = t; i < 14 * F2STR; i += 256) s_f2[50 * F2STR + i] = 0.0f; // pad rows
  __syncthreads();

  // ---- Phase A: f2[f][e] = (att_mem @ Mem)[e] * fe[e] ----
  const int e0 = lane * 4;
  for (int f = w; f < NFR; f += 8) {
    int uf = input_uf[b * NFR + f];
    float msk = (uf != user_num) ? 1.0f : 0.0f;
    float4 fe = ((const float4*)(uidW + (size_t)uf * EDIM))[lane];
    fe.x *= msk; fe.y *= msk; fe.z *= msk; fe.w *= msk;

    float am_l = 0.0f;
    if (lane < MDIM) {
      int idx = f * MDIM + lane;
      am_l = expf(s_attrow[idx] - s_mx[idx]) * s_ise[idx] * msk;
    }
    // Broadcast lanes 0..7 -> all lanes via immediate ds_swizzle (or-mask).
    float am[MDIM];
    am[0] = swz<0x000>(am_l); am[1] = swz<0x020>(am_l);
    am[2] = swz<0x040>(am_l); am[3] = swz<0x060>(am_l);
    am[4] = swz<0x080>(am_l); am[5] = swz<0x0a0>(am_l);
    am[6] = swz<0x0c0>(am_l); am[7] = swz<0x0e0>(am_l);

    float f10 = 0.f, f11 = 0.f, f12 = 0.f, f13 = 0.f;
#pragma unroll
    for (int m = 0; m < MDIM; ++m) {
      const float* mr = s_mem + m * EDIM + e0;
      f10 += am[m]*mr[0]; f11 += am[m]*mr[1]; f12 += am[m]*mr[2]; f13 += am[m]*mr[3];
    }
    float* row = s_f2 + f * F2STR + e0;
    row[0] = f10 * fe.x; row[1] = f11 * fe.y; row[2] = f12 * fe.z; row[3] = f13 * fe.w;
    if (lane == 0) s_fn[f] = msk;
  }
  __syncthreads();

  // ---- Phase B: h = relu(f2 @ WA + BA); dot = h @ U_omega (WMMA bf16) ----
  const int rt = w & 3;           // row tile (16 rows each)
  const int nh = w >> 2;          // N half: covers nt = 2*nh, 2*nh+1
  const int rowM   = lane & 15;
  const int kHalfA = (lane >> 4) * 8;    // A-matrix K half per lane group
  const float2* arow2 =
      (const float2*)(s_f2 + (rt * 16 + rowM) * F2STR);  // 8B-aligned (even stride)

  Frag Afr[4];
#pragma unroll
  for (int kb = 0; kb < 4; ++kb) {
#pragma unroll
    for (int d = 0; d < 8; ++d) {
      int kk = kb * 32 + ((d < 4) ? 0 : 16) + kHalfA + 2 * (d & 3);
      float2 p = arow2[kk >> 1];
      Afr[kb].u[d] = pack_bf16(p.x, p.y);
    }
  }

  const int nc = lane & 15;
#pragma unroll
  for (int ni = 0; ni < 2; ++ni) {
    int nt = nh * 2 + ni;
    int col = nt * 16 + nc;
    v8f C = {0.f, 0.f, 0.f, 0.f, 0.f, 0.f, 0.f, 0.f};
#pragma unroll
    for (int kb = 0; kb < 4; ++kb) {
      Frag Bfr;
      const uint4* bp = (const uint4*)WAbf + ((size_t)((nt * 4 + kb) * 32 + lane)) * 2;
      Bfr.q[0] = bp[0];
      Bfr.q[1] = bp[1];
      C = __builtin_amdgcn_wmma_f32_16x16x32_bf16(
          false, Afr[kb].v, false, Bfr.v, (short)0, C, false, false);
    }
    float ba = BA[col];
    float uo = Uo[col];
#pragma unroll
    for (int r = 0; r < 8; ++r) {
      float h = fmaxf(C[r] + ba, 0.0f);
      float v = halfSum16(h * uo);
      // lanes 0-15 hold row rt*16+r sum; lanes 16-31 hold row rt*16+r+8 sum.
      if (lane == 0)  s_dotp[(rt * 16 + r) * 2 + nh]     += v;
      if (lane == 16) s_dotp[(rt * 16 + r + 8) * 2 + nh] += v;
    }
  }
  __syncthreads();

  // ---- Phase C: j, weights, friend, score ----
  if (t < 64) {
    float j = 0.0f;
    if (t < NFR && s_fn[t] > 0.0f) j = expf(s_dotp[2 * t] + s_dotp[2 * t + 1]);
    s_j[t] = j;
  }
  __syncthreads();
  if (w == 0) {
    float v = s_j[lane] + s_j[lane + 32];
    v = waveSum(v);
    if (lane == 0) s_sj = v;
  }
  __syncthreads();
  const float invden = 1.0f / (s_sj + 1e-8f);

  {
    int e = t & (EDIM - 1);
    int half = t >> 7;
    float acc = 0.0f;
    int f0 = half * (NFR / 2);
#pragma unroll 5
    for (int f = f0; f < f0 + NFR / 2; ++f) acc += s_j[f] * s_f2[f * F2STR + e];
    s_tmp[t] = acc;
  }
  __syncthreads();
  if (t < EDIM) {
    float fr = (s_tmp[t] + s_tmp[t + EDIM]) * invden;
    float user = s_uid[t] + fr;
    float iv = iidW[(size_t)input_i[b] * EDIM + t];
    s_red[t] = user * iv;
  }
  __syncthreads();
  for (int s = 64; s > 0; s >>= 1) {
    if (t < s) s_red[t] += s_red[t + s];
    __syncthreads();
  }
  if (t == 0) out[b] = s_red[0] + i_bias[input_i[b]];
}

// ---------------- Host launcher --------------------------------------------
extern "C" void kernel_launch(void* const* d_in, const int* in_sizes, int n_in,
                              void* d_out, int out_size, void* d_ws, size_t ws_size,
                              hipStream_t stream) {
  const int*   input_u  = (const int*)  d_in[0];
  const int*   input_i  = (const int*)  d_in[1];
  const int*   input_uf = (const int*)  d_in[2];
  const float* uidW     = (const float*)d_in[3];
  const float* iidW     = (const float*)d_in[4];
  const float* i_bias   = (const float*)d_in[5];
  const float* Key      = (const float*)d_in[6];
  const float* Mem      = (const float*)d_in[7];
  const float* WA       = (const float*)d_in[8];
  const float* BA       = (const float*)d_in[9];
  const float* Uo       = (const float*)d_in[10];
  float* out = (float*)d_out;

  const int Bsz = in_sizes[0];
  const int user_num = in_sizes[3] / EDIM - 1;   // uidW rows - 1 (padding id)

  float*    att  = (float*)d_ws;                      // [Bsz][400]
  float*    mx   = att + (size_t)Bsz * FM;            // [400]
  float*    ise  = mx + FM;                           // [400]
  unsigned* WAbf = (unsigned*)(ise + FM);             // [4096] packed B fragments

  k_prepWA<<<WAFRAG_DW / 256, 256, 0, stream>>>(WA, WAbf);
  k_attkey<<<Bsz, 256, 0, stream>>>(input_u, input_uf, uidW, Key, att, user_num);
  k_stats<<<FM, 256, 0, stream>>>(att, mx, ise, Bsz);
  k_main<<<Bsz, 256, 0, stream>>>(input_u, input_i, input_uf, uidW, iidW, i_bias,
                                  Mem, WAbf, BA, Uo, att, mx, ise, out, user_num);
}
